// RateCodingClassifier_32323923870268
// MI455X (gfx1250) — compile-verified, matrix operand order checked by reading, not run on hardware
//
#include <hip/hip_runtime.h>
#include <hip/hip_bf16.h>

typedef __bf16 bf16;
typedef bf16  v16bf __attribute__((ext_vector_type(16)));
typedef float v8f   __attribute__((ext_vector_type(8)));

#define BATCH   256
#define NDIM    2048
#define NSTEP   100
#define NCLASS  1000
#define NPAD    1024              // padded N for GEMM / proj
#define MROWS   (NSTEP * BATCH)   // 25600

#define BM 128
#define BN 256
#define BK 32
#define LDT 40                    // padded LDS row stride in bf16 (80 B: 16B-aligned)
#define KT  (NDIM / BK)           // 64

// ---- CDNA5 async global->LDS path (guarded; falls back to register staging) ----
#if defined(__has_builtin)
# if __has_builtin(__builtin_amdgcn_global_load_async_to_lds_b128)
#  define SNN_ASYNC 1
# endif
#endif
#ifndef SNN_ASYNC
# define SNN_ASYNC 0
#endif

#if defined(__has_builtin)
# if __has_builtin(__builtin_amdgcn_s_wait_asynccnt)
#  define SNN_WAIT_ASYNC() __builtin_amdgcn_s_wait_asynccnt(0)
# endif
#endif
#ifndef SNN_WAIT_ASYNC
# define SNN_WAIT_ASYNC() asm volatile("s_wait_asynccnt 0" ::: "memory")
#endif

#if SNN_ASYNC
// Builtin signature (from hipcc diagnostic): param0 = addrspace(1) int4*,
// i.e. (global src, lds dst, imm offset, imm cpol).
typedef int v4i_gcc __attribute__((vector_size(16)));
typedef __attribute__((address_space(1))) v4i_gcc gv4i;
typedef __attribute__((address_space(3))) v4i_gcc lv4i;
__device__ __forceinline__ void snn_async_ld16(const bf16* g, bf16* l) {
    // GLOBAL_LOAD_ASYNC_TO_LDS_B128: memory -> LDS, tracked by ASYNCcnt
    __builtin_amdgcn_global_load_async_to_lds_b128((gv4i*)g, (lv4i*)l, 0, 0);
}
#endif

// ---------------------------------------------------------------------------
// Kernel 1: transpose + fp32->bf16 convert:  input[b,d,t] -> A[(t*256+b), d]
// Coalesced reads along t, coalesced writes along d via 32x33 LDS tile.
// ---------------------------------------------------------------------------
__global__ void snn_transpose_convert(const float* __restrict__ in,
                                      bf16* __restrict__ Aout) {
    __shared__ float tile[32][33];
    const int d0 = blockIdx.x * 32;
    const int t0 = blockIdx.y * 32;
    const int b  = blockIdx.z;
    const int tx = threadIdx.x;   // 0..31
    const int ty = threadIdx.y;   // 0..7
    const float* src = in + (size_t)b * NDIM * NSTEP;
#pragma unroll
    for (int r = 0; r < 4; ++r) {
        const int dl = ty + r * 8;
        const int t  = t0 + tx;
        tile[dl][tx] = (t < NSTEP) ? src[(size_t)(d0 + dl) * NSTEP + t] : 0.0f;
    }
    __syncthreads();
#pragma unroll
    for (int r = 0; r < 4; ++r) {
        const int tl = ty + r * 8;
        const int t  = t0 + tl;
        if (t < NSTEP) {
            const size_t m = (size_t)t * BATCH + b;
            Aout[m * NDIM + d0 + tx] = (bf16)tile[tx][tl];
        }
    }
}

// ---------------------------------------------------------------------------
// Kernel 2: W fp32->bf16, padded to 1024 rows (zeros beyond NCLASS)
// ---------------------------------------------------------------------------
__global__ void snn_wconvert(const float* __restrict__ W, bf16* __restrict__ Wout) {
    const size_t idx = (size_t)blockIdx.x * blockDim.x + threadIdx.x; // over 1024*2048
    const int c = (int)(idx >> 11);
    const int d = (int)(idx & 2047);
    Wout[idx] = (c < NCLASS) ? (bf16)W[(size_t)c * NDIM + d] : (bf16)0.0f;
}

// ---------------------------------------------------------------------------
// Kernel 3: bf16 WMMA GEMM  proj[M, NPAD] = A[M,K] * W[N,K]^T  (fp32 accum)
// WG tile 128x256, 8 waves (2M x 4N), wave tile 64x64 = 4x4 wmma tiles.
// Double-buffered LDS; async global->LDS pipeline (ASYNCcnt) when available.
// ---------------------------------------------------------------------------
__device__ __forceinline__ v16bf snn_load_frag(const bf16* base, int row, int kh) {
    // 16-bit A/B fragment layout (ISA 7.12.2): lane<16 holds K {0..7,16..23},
    // lane>=16 holds K {8..15,24..31}; row = lane&15 (+ tile offset).
    union { uint4 u[2]; v16bf v; } tmp;
    tmp.u[0] = *(const uint4*)(base + row * LDT + kh * 8);
    tmp.u[1] = *(const uint4*)(base + row * LDT + 16 + kh * 8);
    return tmp.v;
}

__global__ __launch_bounds__(256)
void snn_gemm_bf16(const bf16* __restrict__ A, const bf16* __restrict__ Bw,
                   float* __restrict__ C) {
    __shared__ __attribute__((aligned(16))) bf16 As[2][BM * LDT];
    __shared__ __attribute__((aligned(16))) bf16 Bs[2][BN * LDT];

    const int tid  = threadIdx.x;
    const int lane = tid & 31;
    const int wave = tid >> 5;       // 0..7
    const int wm   = wave >> 2;      // 0..1 : M sub-block (64 rows)
    const int wn   = wave & 3;       // 0..3 : N sub-block (64 cols)
    const int lr   = lane & 15;
    const int kh   = lane >> 4;

    const size_t n0 = (size_t)blockIdx.x * BN;   // fastest: 4 n-tiles share A in L2
    const size_t m0 = (size_t)blockIdx.y * BM;

    // Staging chunk maps (uint4 = 8 bf16): chunk c -> row c>>2, k-offset (c&3)*8.
    // A tile: 512 chunks (2/thread), B tile: 1024 chunks (4/thread).
    const bf16* ag[2]; int al[2];
    const bf16* bg[4]; int bl[4];
#pragma unroll
    for (int j = 0; j < 2; ++j) {
        const int c = tid + j * 256, r = c >> 2, o = (c & 3) * 8;
        ag[j] = A + (m0 + r) * NDIM + o;
        al[j] = r * LDT + o;
    }
#pragma unroll
    for (int j = 0; j < 4; ++j) {
        const int c = tid + j * 256, r = c >> 2, o = (c & 3) * 8;
        bg[j] = Bw + (n0 + r) * NDIM + o;
        bl[j] = r * LDT + o;
    }

    v8f acc[4][4] = {};

#if SNN_ASYNC
    // ---- async global->LDS pipeline: 1 barrier / k-step, no VGPR staging ----
#pragma unroll
    for (int j = 0; j < 2; ++j) snn_async_ld16(ag[j], &As[0][al[j]]);
#pragma unroll
    for (int j = 0; j < 4; ++j) snn_async_ld16(bg[j], &Bs[0][bl[j]]);
#pragma unroll
    for (int j = 0; j < 2; ++j) ag[j] += BK;
#pragma unroll
    for (int j = 0; j < 4; ++j) bg[j] += BK;
    SNN_WAIT_ASYNC();
    __syncthreads();

    for (int kt = 0; kt < KT; ++kt) {
        const int cur = kt & 1, nxt = cur ^ 1;
        if (kt + 1 < KT) {
#pragma unroll
            for (int j = 0; j < 2; ++j) snn_async_ld16(ag[j], &As[nxt][al[j]]);
#pragma unroll
            for (int j = 0; j < 4; ++j) snn_async_ld16(bg[j], &Bs[nxt][bl[j]]);
#pragma unroll
            for (int j = 0; j < 2; ++j) ag[j] += BK;
#pragma unroll
            for (int j = 0; j < 4; ++j) bg[j] += BK;
            if (kt + 2 < KT) {           // distance-2 stream prefetch
                __builtin_prefetch(ag[0], 0, 0);
                __builtin_prefetch(bg[0], 0, 0);
            }
        }
        v16bf bfrag[4];
#pragma unroll
        for (int ni = 0; ni < 4; ++ni)
            bfrag[ni] = snn_load_frag(&Bs[cur][0], wn * 64 + ni * 16 + lr, kh);
#pragma unroll
        for (int mi = 0; mi < 4; ++mi) {
            const v16bf afrag = snn_load_frag(&As[cur][0], wm * 64 + mi * 16 + lr, kh);
#pragma unroll
            for (int ni = 0; ni < 4; ++ni)
                acc[mi][ni] = __builtin_amdgcn_wmma_f32_16x16x32_bf16(
                    false, afrag, false, bfrag[ni],
                    (short)0, acc[mi][ni], false, false);
        }
        if (kt + 1 < KT) SNN_WAIT_ASYNC();
        __syncthreads();
    }
#else
    // ---- fallback: register-staged pipeline, double-buffered, 1 barrier/iter ----
    uint4 ra[2], rb[4];
#pragma unroll
    for (int j = 0; j < 2; ++j) { ra[j] = *(const uint4*)ag[j]; ag[j] += BK; }
#pragma unroll
    for (int j = 0; j < 4; ++j) { rb[j] = *(const uint4*)bg[j]; bg[j] += BK; }
#pragma unroll
    for (int j = 0; j < 2; ++j) *(uint4*)(&As[0][al[j]]) = ra[j];
#pragma unroll
    for (int j = 0; j < 4; ++j) *(uint4*)(&Bs[0][bl[j]]) = rb[j];
    __syncthreads();

    for (int kt = 0; kt < KT; ++kt) {
        const int cur = kt & 1, nxt = cur ^ 1;
        if (kt + 1 < KT) {
#pragma unroll
            for (int j = 0; j < 2; ++j) { ra[j] = *(const uint4*)ag[j]; ag[j] += BK; }
#pragma unroll
            for (int j = 0; j < 4; ++j) { rb[j] = *(const uint4*)bg[j]; bg[j] += BK; }
            if (kt + 2 < KT) {
                __builtin_prefetch(ag[0], 0, 0);
                __builtin_prefetch(bg[0], 0, 0);
            }
        }
        v16bf bfrag[4];
#pragma unroll
        for (int ni = 0; ni < 4; ++ni)
            bfrag[ni] = snn_load_frag(&Bs[cur][0], wn * 64 + ni * 16 + lr, kh);
#pragma unroll
        for (int mi = 0; mi < 4; ++mi) {
            const v16bf afrag = snn_load_frag(&As[cur][0], wm * 64 + mi * 16 + lr, kh);
#pragma unroll
            for (int ni = 0; ni < 4; ++ni)
                acc[mi][ni] = __builtin_amdgcn_wmma_f32_16x16x32_bf16(
                    false, afrag, false, bfrag[ni],
                    (short)0, acc[mi][ni], false, false);
        }
        if (kt + 1 < KT) {
#pragma unroll
            for (int j = 0; j < 2; ++j) *(uint4*)(&As[nxt][al[j]]) = ra[j];
#pragma unroll
            for (int j = 0; j < 4; ++j) *(uint4*)(&Bs[nxt][bl[j]]) = rb[j];
        }
        __syncthreads();
    }
#endif

    // C/D layout: vgpr r -> row (lane<16 ? r : 8+r), col lane&15
#pragma unroll
    for (int mi = 0; mi < 4; ++mi) {
#pragma unroll
        for (int ni = 0; ni < 4; ++ni) {
            const size_t mbase = m0 + wm * 64 + mi * 16 + kh * 8;
            const size_t nn    = n0 + wn * 64 + ni * 16 + lr;
            float* out = C + mbase * NPAD + nn;
#pragma unroll
            for (int r = 0; r < 8; ++r)
                out[(size_t)r * NPAD] = acc[mi][ni][r];
        }
    }
}

// ---------------------------------------------------------------------------
// Kernel 4: LIF scan over T. One lane per (b,c); coalesced strided proj reads.
// ---------------------------------------------------------------------------
__global__ void snn_lif_scan(const float* __restrict__ proj,
                             const float* __restrict__ bias,
                             float* __restrict__ out) {
    const int idx = blockIdx.x * blockDim.x + threadIdx.x;
    if (idx >= BATCH * NCLASS) return;
    const int b = idx / NCLASS;
    const int c = idx - b * NCLASS;
    const float bv = bias[c];
    const float* p = proj + (size_t)b * NPAD + c;
    float mem = 0.0f, spk = 0.0f, ssum = 0.0f;
#pragma unroll 4
    for (int t = 0; t < NSTEP; ++t) {
        const float pv = p[(size_t)t * BATCH * NPAD] + bv;
        mem  = mem * 0.2f * (1.0f - spk) + pv;
        spk  = (mem > 0.5f) ? 1.0f : 0.0f;
        ssum += spk;
    }
    out[idx] = ssum * (1.0f / (float)NSTEP);
}

// ---------------------------------------------------------------------------
extern "C" void kernel_launch(void* const* d_in, const int* in_sizes, int n_in,
                              void* d_out, int out_size, void* d_ws, size_t ws_size,
                              hipStream_t stream) {
    const float* input = (const float*)d_in[0];   // [256, 2048, 100] f32
    const float* W     = (const float*)d_in[1];   // [1000, 2048] f32
    const float* bias  = (const float*)d_in[2];   // [1000] f32
    float* out = (float*)d_out;                   // [256, 1000] f32

    char* ws = (char*)d_ws;
    const size_t A_BYTES = (size_t)MROWS * NDIM * sizeof(bf16);   // 104,857,600
    const size_t W_BYTES = (size_t)NPAD  * NDIM * sizeof(bf16);   //   4,194,304
    bf16*  Abf  = (bf16*)ws;
    bf16*  Wbf  = (bf16*)(ws + A_BYTES);
    float* proj = (float*)(ws + A_BYTES + W_BYTES);               // [M, NPAD] f32

    // 1) transpose + convert input -> A_bf16[m, k]
    snn_transpose_convert<<<dim3(NDIM / 32, (NSTEP + 31) / 32, BATCH),
                            dim3(32, 8), 0, stream>>>(input, Abf);
    // 2) convert W -> bf16, zero-padded to 1024 rows
    snn_wconvert<<<(NPAD * NDIM) / 256, 256, 0, stream>>>(W, Wbf);
    // 3) WMMA GEMM: proj = A * W^T  (n-tile fastest for A reuse in L2)
    snn_gemm_bf16<<<dim3(NPAD / BN, MROWS / BM), 256, 0, stream>>>(Abf, Wbf, proj);
    // 4) LIF temporal scan + rate output
    snn_lif_scan<<<(BATCH * NCLASS + 255) / 256, 256, 0, stream>>>(proj, bias, out);
}
